// MetaGIN_71975061946738
// MI455X (gfx1250) — matrix operand order repaired and checked by reading, not compiled, not run on hardware
//
#include <hip/hip_runtime.h>
#include <hip/hip_bf16.h>

// ---------------------------------------------------------------------------
// MetaGIN fused layer for MI455X (gfx1250, wave32, WMMA).
// All GEMMs routed through v_wmma_f32_16x16x32_bf16 (bf16 A/B, f32 accum).
// 16x64 tiles per wave: one A fragment feeds 4 WMMAs per K-step so L2
// latency is hidden behind matrix math instead of s_wait_loadcnt stalls.
// ---------------------------------------------------------------------------

typedef __bf16 bf16_t;
typedef __attribute__((ext_vector_type(16))) __bf16 v16bf;
typedef __attribute__((ext_vector_type(8)))  __bf16 v8bf;
typedef __attribute__((ext_vector_type(4)))  __bf16 v4bf;
typedef __attribute__((ext_vector_type(8)))  float  v8f;

#define WIDTH  256
#define NHEAD  8
#define DHEAD  32
#define EPS_GN 1e-5f

static __device__ __forceinline__ bf16_t f2bf(float f) { return (bf16_t)f; }

static __device__ __forceinline__ v16bf cat8(v8bf lo, v8bf hi) {
    return __builtin_shufflevector(lo, hi,
        0,1,2,3,4,5,6,7,8,9,10,11,12,13,14,15);
}

// ISA 7.12.2 16-bit B-matrix 32x16: lane = column N (mod 16);
// per-lane K runs {0..7,16..23} (lanes 0-15) / {8..15,24..31} (lanes 16-31) -> two 16B runs.
static __device__ __forceinline__ v16bf ld_frag_b16(const bf16_t* base, int lane) {
    const bf16_t* p = base + ((lane >> 4) << 3);
    return cat8(*(const v8bf*)p, *(const v8bf*)(p + 16));
}
// 16-bit A-matrix 16x32: lane = row M (mod 16); per-lane K runs
// {0..7,16..23} (lanes 0-15) / {8..15,24..31} (lanes 16-31) -> two 16B runs.
static __device__ __forceinline__ v16bf ld_frag_a_bf(const bf16_t* base, int lane) {
    const bf16_t* p = base + ((lane >> 4) << 3);
    return cat8(*(const v8bf*)p, *(const v8bf*)(p + 16));
}
// Same A layout sourced from fp32 memory: float4 loads + pack-convert.
static __device__ __forceinline__ v16bf ld_frag_a_f32(const float* base, int lane) {
    const float* p = base + ((lane >> 4) << 3);
    float4 a0 = *(const float4*)(p);
    float4 a1 = *(const float4*)(p + 4);
    float4 a2 = *(const float4*)(p + 16);
    float4 a3 = *(const float4*)(p + 20);
    v16bf r;
    r[0] = f2bf(a0.x); r[1] = f2bf(a0.y); r[2]  = f2bf(a0.z); r[3]  = f2bf(a0.w);
    r[4] = f2bf(a1.x); r[5] = f2bf(a1.y); r[6]  = f2bf(a1.z); r[7]  = f2bf(a1.w);
    r[8] = f2bf(a2.x); r[9] = f2bf(a2.y); r[10] = f2bf(a2.z); r[11] = f2bf(a2.w);
    r[12]= f2bf(a3.x); r[13]= f2bf(a3.y); r[14] = f2bf(a3.z); r[15] = f2bf(a3.w);
    return r;
}

static __device__ __forceinline__ v8f wmma_bf16(v16bf a, v16bf b, v8f c) {
    return __builtin_amdgcn_wmma_f32_16x16x32_bf16(
        false, a, false, b, (short)0, c, false, false);
}

// ---------------------------------------------------------------------------
// Kernel 0: fp32 -> bf16 weight conversion (vectorized 4/thread)
// ---------------------------------------------------------------------------
__global__ void convert_weights(const float4* __restrict__ src,
                                v4bf* __restrict__ dst, int n4) {
    int i = blockIdx.x * blockDim.x + threadIdx.x;
    if (i < n4) {
        float4 v = src[i];
        v4bf o;
        o[0] = f2bf(v.x); o[1] = f2bf(v.y); o[2] = f2bf(v.z); o[3] = f2bf(v.w);
        dst[i] = o;
    }
}

// ---------------------------------------------------------------------------
// Kernel 1: xs = x @ W_src^T, xt = x @ W_tgt^T.
// One wave per 16x64 output tile (4 accumulators, A reused 4x per K-step);
// K=256 in 8 steps -> 32 WMMAs per wave. LDS-transposed b128 store epilogue.
// grid = (ceil(N/16), WIDTH/64, 2)  block = 32
// ---------------------------------------------------------------------------
__global__ __launch_bounds__(32)
void node_dual_gemm(const float* __restrict__ x,
                    const bf16_t* __restrict__ Wsrc,
                    const bf16_t* __restrict__ Wtgt,
                    float* __restrict__ xs, float* __restrict__ xt, int nrows) {
    __shared__ __align__(16) float t[16][64];
    const bf16_t* W   = blockIdx.z ? Wtgt : Wsrc;
    float*        out = blockIdx.z ? xt   : xs;
    const int lane = threadIdx.x;
    const int m0 = blockIdx.x * 16;
    const int n0 = blockIdx.y * 64;
    int m = m0 + (lane & 15); if (m >= nrows) m = nrows - 1;   // clamp, no EXEC change

    const float* arow = x + (size_t)m * WIDTH;
    const bf16_t* brow[4];
#pragma unroll
    for (int tt = 0; tt < 4; ++tt)
        brow[tt] = W + (size_t)(n0 + tt * 16 + (lane & 15)) * WIDTH;

    v8f acc0 = {}, acc1 = {}, acc2 = {}, acc3 = {};
#pragma unroll
    for (int kc = 0; kc < WIDTH; kc += 32) {
        v16bf a = ld_frag_a_f32(arow + kc, lane);       // 1 A load ...
        acc0 = wmma_bf16(a, ld_frag_b16(brow[0] + kc, lane), acc0);  // ... 4 WMMAs
        acc1 = wmma_bf16(a, ld_frag_b16(brow[1] + kc, lane), acc1);
        acc2 = wmma_bf16(a, ld_frag_b16(brow[2] + kc, lane), acc2);
        acc3 = wmma_bf16(a, ld_frag_b16(brow[3] + kc, lane), acc3);
    }

    // C/D layout: lane holds column n, rows mb..mb+7 -> transpose via LDS
    {
        const int nn = lane & 15, mb = (lane >> 4) << 3;
#pragma unroll
        for (int i = 0; i < 8; ++i) {
            t[mb + i][nn]      = acc0[i];
            t[mb + i][16 + nn] = acc1[i];
            t[mb + i][32 + nn] = acc2[i];
            t[mb + i][48 + nn] = acc3[i];
        }
    }
    __syncthreads();
    // 16 rows x 16 float4 = 256 float4; 8 coalesced b128 stores per lane
    const bool full = (m0 + 16 <= nrows);               // uniform
#pragma unroll
    for (int i = 0; i < 8; ++i) {
        int f = lane + 32 * i;
        int r = f >> 4, c4 = f & 15;
        float4 v = *(const float4*)&t[r][c4 * 4];
        if (full || (m0 + r < nrows))
            *(float4*)(out + (size_t)(m0 + r) * WIDTH + n0 + c4 * 4) = v;
    }
}

// ---------------------------------------------------------------------------
// Kernel 2: fused per-edge pipeline. One wave handles 16 edges:
//   gather xs/xt rows -> GroupNorm (wave32 shfl reductions, D==32) ->
//   embedding-bag gate bias -> gate/val grouped WMMAs (K=32 per head) ->
//   post GEMM 16x256 x 256x256 (4 N-tiles per K-sweep) -> atomic scatter.
// ---------------------------------------------------------------------------
__global__ __launch_bounds__(32)
void edge_fused(const float* __restrict__ xs, const float* __restrict__ xt,
                const int* __restrict__ edge_idx, const int* __restrict__ edge_attr,
                const float* __restrict__ emb,
                const bf16_t* __restrict__ Wg, const bf16_t* __restrict__ Wv,
                const bf16_t* __restrict__ Wp,
                float* __restrict__ agg, int E) {
    __shared__ __align__(16) float  s_xx[16][WIDTH];   // 16KB; reused as bf16 act
    __shared__ __align__(16) bf16_t s_xn[16][WIDTH];   // 8KB
    __shared__ __align__(16) bf16_t s_gin[16][WIDTH];  // 8KB
    __shared__ int   s_src[16], s_tgt[16], s_attr[16][3];
    __shared__ float s_icnt[16];
    bf16_t (*s_act)[WIDTH] = (bf16_t (*)[WIDTH])s_xx;  // alias (xx dead by then)

    const int lane = threadIdx.x;
    const int e0 = blockIdx.x * 16;
    const bool full = (e0 + 16 <= E);                  // uniform

    // --- stage 1: edge metadata -------------------------------------------
    if (lane < 16) {
        int e = e0 + lane;
        int clamped = (e < E) ? e : (E - 1);
        s_src[lane] = edge_idx[clamped];                 // edge_idx[0][e]
        s_tgt[lane] = edge_idx[(size_t)E + clamped];     // edge_idx[1][e]
        int a0 = edge_attr[clamped * 3 + 0];
        int a1 = edge_attr[clamped * 3 + 1];
        int a2 = edge_attr[clamped * 3 + 2];
        s_attr[lane][0] = a0; s_attr[lane][1] = a1; s_attr[lane][2] = a2;
        int cnt = (a0 != 0) + (a1 != 0) + (a2 != 0);
        s_icnt[lane] = 1.0f / (float)(cnt > 0 ? cnt : 1);
    }
    __syncthreads();

    // --- stage 2: gather + combine xx = xs[src] + xt[tgt] -----------------
#pragma unroll 4
    for (int e = 0; e < 16; ++e) {
        const float4* ps = (const float4*)(xs + (size_t)s_src[e] * WIDTH);
        const float4* pt = (const float4*)(xt + (size_t)s_tgt[e] * WIDTH);
#pragma unroll
        for (int r = 0; r < 2; ++r) {
            int c4 = lane + r * 32;                      // 64 float4 per row
            float4 a = ps[c4], b = pt[c4];
            float4* d = (float4*)&s_xx[e][c4 * 4];
            d->x = a.x + b.x; d->y = a.y + b.y;
            d->z = a.z + b.z; d->w = a.w + b.w;
        }
    }
    __syncthreads();

    // --- stage 3: GroupNorm + embedding-bag bias --------------------------
    for (int e = 0; e < 16; ++e) {
        const int a0 = s_attr[e][0], a1 = s_attr[e][1], a2 = s_attr[e][2];
        const float icnt = s_icnt[e];
#pragma unroll
        for (int h = 0; h < NHEAD; ++h) {
            int c = h * DHEAD + lane;                    // lane == d-index, D == 32
            float v = s_xx[e][c];
            float sum = v, sq = v * v;
#pragma unroll
            for (int mm = 16; mm >= 1; mm >>= 1) {
                sum += __shfl_xor(sum, mm, 32);
                sq  += __shfl_xor(sq,  mm, 32);
            }
            float mean = sum * (1.0f / 32.0f);
            float var  = sq * (1.0f / 32.0f) - mean * mean;
            float xn = (v - mean) * rsqrtf(var + EPS_GN);
            s_xn[e][c] = f2bf(xn);
            float gb = 0.f;
            if (a0) gb += emb[a0 * WIDTH + c];
            if (a1) gb += emb[a1 * WIDTH + c];
            if (a2) gb += emb[a2 * WIDTH + c];
            s_gin[e][c] = f2bf(xn + gb * icnt);
        }
    }
    __syncthreads();

    // --- stage 4: gate/val grouped matmuls (one WMMA per 16-col tile) -----
    for (int h = 0; h < NHEAD; ++h) {
        const int m = lane & 15;
        v16bf ag = ld_frag_a_bf(&s_gin[m][h * DHEAD], lane);
        v16bf av = ld_frag_a_bf(&s_xn[m][h * DHEAD],  lane);
#pragma unroll
        for (int t = 0; t < 2; ++t) {
            int f = t * 16 + (lane & 15);
            v16bf bg = ld_frag_b16(Wg + (size_t)(h * DHEAD + f) * DHEAD, lane);
            v16bf bv = ld_frag_b16(Wv + (size_t)(h * DHEAD + f) * DHEAD, lane);
            v8f zero = {};
            v8f g  = wmma_bf16(ag, bg, zero);
            v8f vv = wmma_bf16(av, bv, zero);
            const int n  = lane & 15;
            const int mb = (lane >> 4) << 3;
#pragma unroll
            for (int i = 0; i < 8; ++i) {
                float gg = g[i] > 0.f ? g[i] : 0.f;      // relu(gate) * val
                s_act[mb + i][h * DHEAD + t * 16 + n] = f2bf(gg * vv[i]);
            }
        }
    }
    __syncthreads();

    // --- stage 5: out_e = act @ W_post^T, 4 N-tiles per K-sweep -----------
    const int m  = lane & 15;
    const int mb = (lane >> 4) << 3;
    for (int ntg = 0; ntg < 4; ++ntg) {
        const bf16_t* wr0 = Wp + (size_t)((ntg * 4 + 0) * 16 + m) * WIDTH;
        const bf16_t* wr1 = Wp + (size_t)((ntg * 4 + 1) * 16 + m) * WIDTH;
        const bf16_t* wr2 = Wp + (size_t)((ntg * 4 + 2) * 16 + m) * WIDTH;
        const bf16_t* wr3 = Wp + (size_t)((ntg * 4 + 3) * 16 + m) * WIDTH;
        __builtin_prefetch(wr0 + 64 * WIDTH, 0, 1);      // next group's rows
        v8f acc0 = {}, acc1 = {}, acc2 = {}, acc3 = {};
#pragma unroll
        for (int kc = 0; kc < WIDTH; kc += 32) {
            v16bf a = ld_frag_a_bf(&s_act[m][kc], lane); // 1 LDS A load ...
            acc0 = wmma_bf16(a, ld_frag_b16(wr0 + kc, lane), acc0);  // 4 WMMAs
            acc1 = wmma_bf16(a, ld_frag_b16(wr1 + kc, lane), acc1);
            acc2 = wmma_bf16(a, ld_frag_b16(wr2 + kc, lane), acc2);
            acc3 = wmma_bf16(a, ld_frag_b16(wr3 + kc, lane), acc3);
        }
#pragma unroll
        for (int i = 0; i < 8; ++i) {
            if (full || (e0 + mb + i < E)) {
                size_t row = (size_t)s_tgt[mb + i] * WIDTH;
                atomicAdd(&agg[row + (ntg * 4 + 0) * 16 + m], acc0[i]);
                atomicAdd(&agg[row + (ntg * 4 + 1) * 16 + m], acc1[i]);
                atomicAdd(&agg[row + (ntg * 4 + 2) * 16 + m], acc2[i]);
                atomicAdd(&agg[row + (ntg * 4 + 3) * 16 + m], acc3[i]);
            }
        }
    }
}

// ---------------------------------------------------------------------------
// Kernel 3: out = deg^deg_param * agg
// ---------------------------------------------------------------------------
__global__ void finalize(const float* __restrict__ agg,
                         const float* __restrict__ deg,
                         const float* __restrict__ deg_param,
                         float* __restrict__ out, int N) {
    int node = blockIdx.x;
    int c = threadIdx.x;
    if (node < N) {
        float d = deg[node];
        out[(size_t)node * WIDTH + c] =
            powf(d, deg_param[c]) * agg[(size_t)node * WIDTH + c];
    }
}

// ---------------------------------------------------------------------------
extern "C" void kernel_launch(void* const* d_in, const int* in_sizes, int n_in,
                              void* d_out, int out_size, void* d_ws, size_t ws_size,
                              hipStream_t stream) {
    const float* x         = (const float*)d_in[0];
    const float* deg       = (const float*)d_in[1];
    const int*   edge_idx  = (const int*)  d_in[2];
    const int*   edge_attr = (const int*)  d_in[3];
    const float* W_src     = (const float*)d_in[4];
    const float* W_tgt     = (const float*)d_in[5];
    const float* emb       = (const float*)d_in[6];
    const float* W_gate    = (const float*)d_in[7];
    const float* W_val     = (const float*)d_in[8];
    const float* W_post    = (const float*)d_in[9];
    const float* deg_param = (const float*)d_in[10];
    (void)n_in; (void)out_size; (void)ws_size;

    const int N = in_sizes[0] / WIDTH;
    const int E = in_sizes[2] / 2;
    const size_t nodeBytes = (size_t)N * WIDTH * sizeof(float);
    const int WW  = WIDTH * WIDTH;             // 65536
    const int GVW = NHEAD * DHEAD * DHEAD;     // 8192

    // workspace carve (256B-aligned slabs)
    char* ws = (char*)d_ws;
    size_t off = 0;
    auto carve = [&](size_t bytes) -> char* {
        char* p = ws + off;
        off += (bytes + 255) & ~(size_t)255;
        return p;
    };
    float*  xs      = (float*) carve(nodeBytes);
    float*  xt      = (float*) carve(nodeBytes);
    float*  agg     = (float*) carve(nodeBytes);
    bf16_t* Wsrc_bf = (bf16_t*)carve((size_t)WW  * 2);
    bf16_t* Wtgt_bf = (bf16_t*)carve((size_t)WW  * 2);
    bf16_t* Wp_bf   = (bf16_t*)carve((size_t)WW  * 2);
    bf16_t* Wg_bf   = (bf16_t*)carve((size_t)GVW * 2);
    bf16_t* Wv_bf   = (bf16_t*)carve((size_t)GVW * 2);

    // stage 0: weight conversion + zero the accumulator
    convert_weights<<<(WW/4  + 255) / 256, 256, 0, stream>>>(
        (const float4*)W_src,  (v4bf*)Wsrc_bf, WW / 4);
    convert_weights<<<(WW/4  + 255) / 256, 256, 0, stream>>>(
        (const float4*)W_tgt,  (v4bf*)Wtgt_bf, WW / 4);
    convert_weights<<<(WW/4  + 255) / 256, 256, 0, stream>>>(
        (const float4*)W_post, (v4bf*)Wp_bf,   WW / 4);
    convert_weights<<<(GVW/4 + 255) / 256, 256, 0, stream>>>(
        (const float4*)W_gate, (v4bf*)Wg_bf,   GVW / 4);
    convert_weights<<<(GVW/4 + 255) / 256, 256, 0, stream>>>(
        (const float4*)W_val,  (v4bf*)Wv_bf,   GVW / 4);
    hipMemsetAsync(agg, 0, nodeBytes, stream);

    // stage 1: node-level dual GEMM (16x64 tiles)
    dim3 g1((N + 15) / 16, WIDTH / 64, 2);
    node_dual_gemm<<<g1, 32, 0, stream>>>(x, Wsrc_bf, Wtgt_bf, xs, xt, N);

    // stage 2: fused edge pipeline (16 edges per wave)
    edge_fused<<<(E + 15) / 16, 32, 0, stream>>>(
        xs, xt, edge_idx, edge_attr, emb, Wg_bf, Wv_bf, Wp_bf, agg, E);

    // stage 3: degree rescale
    finalize<<<N, WIDTH, 0, stream>>>(agg, deg, deg_param, (float*)d_out, N);
}
